// RecurrentEncoder_13657996001376
// MI455X (gfx1250) — matrix operand order repaired and checked by reading, not compiled, not run on hardware
//
#include <hip/hip_runtime.h>

// ---------------------------------------------------------------------------
// Problem constants (match reference)
// ---------------------------------------------------------------------------
#define NN     8192
#define TT     48
#define HH     256
#define LL     2
#define NENV   10
#define NIN    16
#define FOURH  1024          // 4*H
#define KCAT   512           // H (input) + H (hidden) concatenated along K

typedef __attribute__((ext_vector_type(16))) __bf16 v16bf;
typedef __attribute__((ext_vector_type(8)))  __bf16 v8bf;
typedef __attribute__((ext_vector_type(8)))  float  v8f;

union ABFrag { v16bf v; v8bf h[2]; };

// ---------------------------------------------------------------------------
// Workspace layout (bytes), all 256B aligned
// ---------------------------------------------------------------------------
#define OFF_WCAT  0u                                   // 2 * 1024 * 512 bf16 = 2 MB
#define OFF_BIAS  (OFF_WCAT + 2u*FOURH*KCAT*2u)        // 2 * 1024 f32
#define OFF_HBF   (OFF_BIAS + 2u*FOURH*4u)             // 2 * N * H bf16 = 8 MB
#define OFF_INP0  (OFF_HBF  + 2u*NN*HH*2u)             // N * H bf16 = 4 MB
#define OFF_GATES (OFF_INP0 + (unsigned)NN*HH*2u)      // N * 4H f32 = 32 MB

// ---------------------------------------------------------------------------
// Zero h/c state (lives in d_out => final output is the state) and bf16 h copy
// ---------------------------------------------------------------------------
__global__ void k_zero_state(float* __restrict__ out, __bf16* __restrict__ hbf) {
    size_t idx = (size_t)blockIdx.x * blockDim.x + threadIdx.x;
    if (idx < (size_t)2 * LL * NN * HH) out[idx] = 0.0f;            // h then c
    if (idx < (size_t)LL * NN * HH)     hbf[idx] = (__bf16)0.0f;
}

// ---------------------------------------------------------------------------
// Pack weights: Wcat[l][m][k] = (k<256 ? W_ih[l][m][k] : W_hh[l][m][k-256]) as bf16
// (k-contiguous => B-fragment loads are contiguous 16B per lane)
// Also bias[l][m] = b_ih + b_hh.
// ---------------------------------------------------------------------------
__global__ void k_prep_weights(const float* __restrict__ Wih,
                               const float* __restrict__ Whh,
                               const float* __restrict__ bih,
                               const float* __restrict__ bhh,
                               __bf16* __restrict__ wcat,
                               float*  __restrict__ bias) {
    size_t idx = (size_t)blockIdx.x * blockDim.x + threadIdx.x;   // L*4H*512 total
    size_t l   = idx / ((size_t)FOURH * KCAT);
    size_t rem = idx % ((size_t)FOURH * KCAT);
    size_t m   = rem / KCAT;
    size_t k   = rem % KCAT;
    float v = (k < HH) ? Wih[(l * FOURH + m) * HH + k]
                       : Whh[(l * FOURH + m) * HH + (k - HH)];
    wcat[idx] = (__bf16)v;
    if (idx < (size_t)LL * FOURH) bias[idx] = bih[idx] + bhh[idx];
}

// ---------------------------------------------------------------------------
// Per-timestep input projection: inp0[n][h] = bf16( sum_f W_in[h][f] * feats[n][f][t] )
// One block per n; feats staged through LDS, one thread per h.
// ---------------------------------------------------------------------------
__global__ void k_proj(const float* __restrict__ x,      // [N,T]
                       const float* __restrict__ coords, // [N,2]
                       const float* __restrict__ env,    // [N,NENV,T]
                       const float* __restrict__ areas,  // [N]
                       const float* __restrict__ buv,    // [N,2,T]
                       const float* __restrict__ Win,    // [H,NIN]
                       __bf16* __restrict__ inp0,        // [N,H]
                       int t) {
    __shared__ float feats[NIN];
    int n = blockIdx.x;
    int h = threadIdx.x;
    if (h < NIN) {
        float v;
        if      (h == 0)  v = x[(size_t)n * TT + t];
        else if (h <= 2)  v = coords[(size_t)n * 2 + (h - 1)];
        else if (h <= 12) v = env[((size_t)n * NENV + (h - 3)) * TT + t];
        else if (h == 13) v = areas[n];
        else              v = buv[((size_t)n * 2 + (h - 14)) * TT + t];
        feats[h] = v;
    }
    __syncthreads();
    float acc = 0.0f;
    #pragma unroll
    for (int f = 0; f < NIN; ++f) acc += Win[h * NIN + f] * feats[f];
    inp0[(size_t)n * HH + h] = (__bf16)acc;
}

// ---------------------------------------------------------------------------
// Gate GEMM: G[N,4H] = [A1 | A2] (N x 512, bf16) @ Wcatᵀ   via v_wmma_f32_16x16x32_bf16
//   A1 = timestep input (N x 256), A2 = recurrent h (N x 256)
//   B element B[k,m] = wcat_l[m*512 + k]  (k-contiguous per output unit m)
// Block = 256 threads (8 wave32), tile 128(M) x 128(N); wave tile 32 x 64.
// ---------------------------------------------------------------------------
__global__ void __launch_bounds__(256)
k_gemm_gates(const __bf16* __restrict__ A1,
             const __bf16* __restrict__ A2,
             const __bf16* __restrict__ wcat_l,   // [4H, 512]
             float* __restrict__ gates) {         // [N, 4H]
    const int lane = threadIdx.x & 31;
    const int wv   = threadIdx.x >> 5;
    const int lm   = lane & 15;        // fragment row (A) / column (B,C,D)
    const int lk   = lane >> 4;        // K-half selector

    const int mBase = blockIdx.x * 128 + (wv & 3) * 32;   // sample rows
    const int nBase = blockIdx.y * 128 + (wv >> 2) * 64;  // gate-unit cols

    v8f acc[2][4] = {};

    #pragma unroll
    for (int kk = 0; kk < KCAT / 32; ++kk) {
        const int k0 = kk * 32;
        const __bf16* Abase = (k0 < HH) ? A1 : A2;
        const int kA = (k0 < HH) ? k0 : (k0 - HH);

        // A fragments (16x32): lane lm = row; chunks at kA+lk*8 and +16
        v16bf a[2];
        #pragma unroll
        for (int mi = 0; mi < 2; ++mi) {
            const __bf16* p = Abase + (size_t)(mBase + mi * 16 + lm) * HH + kA + lk * 8;
            ABFrag u;
            u.h[0] = *(const v8bf*)(p);
            u.h[1] = *(const v8bf*)(p + 16);
            a[mi] = u.v;
        }
        // B fragments (32x16): lane lm = column m; chunks at k0+lk*8 and +16
        v16bf b[4];
        #pragma unroll
        for (int ni = 0; ni < 4; ++ni) {
            const __bf16* p = wcat_l + (size_t)(nBase + ni * 16 + lm) * KCAT + k0 + lk * 8;
            ABFrag u;
            u.h[0] = *(const v8bf*)(p);
            u.h[1] = *(const v8bf*)(p + 16);
            b[ni] = u.v;
        }
        #pragma unroll
        for (int mi = 0; mi < 2; ++mi)
            #pragma unroll
            for (int ni = 0; ni < 4; ++ni)
                acc[mi][ni] = __builtin_amdgcn_wmma_f32_16x16x32_bf16(
                    false, a[mi], false, b[ni], (short)0, acc[mi][ni], false, false);
    }

    // D layout: VGPR r -> row r + 8*lk ; lane lm -> column
    #pragma unroll
    for (int mi = 0; mi < 2; ++mi)
        #pragma unroll
        for (int ni = 0; ni < 4; ++ni)
            #pragma unroll
            for (int r = 0; r < 8; ++r) {
                int row = mBase + mi * 16 + r + 8 * lk;
                int col = nBase + ni * 16 + lm;
                gates[(size_t)row * FOURH + col] = acc[mi][ni][r];
            }
}

// ---------------------------------------------------------------------------
// Pointwise LSTM cell: c' = σ(f)*c + σ(i)*tanh(g);  h' = σ(o)*tanh(c')
// Writes f32 h,c (into d_out regions) and bf16 h for the next GEMM.
// ---------------------------------------------------------------------------
__device__ __forceinline__ float sigm(float v) { return 1.0f / (1.0f + __expf(-v)); }

__global__ void k_pointwise(const float* __restrict__ gates,   // [N, 4H]
                            const float* __restrict__ bias_l,  // [4H]
                            float* __restrict__ h_l,           // [N, H]
                            float* __restrict__ c_l,           // [N, H]
                            __bf16* __restrict__ hbf_l) {      // [N, H]
    size_t idx = (size_t)blockIdx.x * blockDim.x + threadIdx.x;  // N*H
    size_t n = idx / HH, j = idx % HH;
    size_t base = n * FOURH;
    float gi = gates[base +          j] + bias_l[j];
    float gf = gates[base +   HH  +  j] + bias_l[HH + j];
    float gg = gates[base + 2*HH  +  j] + bias_l[2*HH + j];
    float go = gates[base + 3*HH  +  j] + bias_l[3*HH + j];
    float c  = c_l[idx];
    float cn = sigm(gf) * c + sigm(gi) * tanhf(gg);
    float hn = sigm(go) * tanhf(cn);
    c_l[idx] = cn;
    h_l[idx] = hn;
    hbf_l[idx] = (__bf16)hn;
}

// ---------------------------------------------------------------------------
// Host orchestration (all on `stream`; graph-capture safe)
// ---------------------------------------------------------------------------
extern "C" void kernel_launch(void* const* d_in, const int* in_sizes, int n_in,
                              void* d_out, int out_size, void* d_ws, size_t ws_size,
                              hipStream_t stream) {
    const float* x      = (const float*)d_in[0];
    const float* coords = (const float*)d_in[1];
    const float* env    = (const float*)d_in[2];
    const float* areas  = (const float*)d_in[3];
    const float* buv    = (const float*)d_in[4];
    const float* Win    = (const float*)d_in[5];
    const float* Wih    = (const float*)d_in[6];
    const float* Whh    = (const float*)d_in[7];
    const float* bih    = (const float*)d_in[8];
    const float* bhh    = (const float*)d_in[9];

    float* out    = (float*)d_out;               // [h(L,N,H) | c(L,N,H)]
    float* h_base = out;
    float* c_base = out + (size_t)LL * NN * HH;

    char* ws = (char*)d_ws;
    __bf16* wcat  = (__bf16*)(ws + OFF_WCAT);
    float*  bias  = (float*) (ws + OFF_BIAS);
    __bf16* hbf   = (__bf16*)(ws + OFF_HBF);
    __bf16* inp0  = (__bf16*)(ws + OFF_INP0);
    float*  gates = (float*) (ws + OFF_GATES);

    // 1) zero recurrent state (h,c in d_out; bf16 h copies)
    k_zero_state<<<(2u * LL * NN * HH + 255) / 256, 256, 0, stream>>>(out, hbf);

    // 2) pack weights to bf16 [Wihᵀ;Whhᵀ] and fold biases
    k_prep_weights<<<(2u * FOURH * KCAT + 255) / 256, 256, 0, stream>>>(
        Wih, Whh, bih, bhh, wcat, bias);

    dim3 gemmGrid(NN / 128, FOURH / 128);        // 64 x 8 blocks
    const unsigned pwBlocks = (NN * HH) / 256;   // 8192

    // 3) sequential timestep loop; within a step the 5 kernels are dependent
    for (int t = 0; t < TT; ++t) {
        k_proj<<<NN, HH, 0, stream>>>(x, coords, env, areas, buv, Win, inp0, t);

        // layer 0: A = [proj_t | h0], W = Wcat0
        k_gemm_gates<<<gemmGrid, 256, 0, stream>>>(inp0, hbf, wcat, gates);
        k_pointwise<<<pwBlocks, 256, 0, stream>>>(gates, bias,
                                                  h_base, c_base, hbf);

        // layer 1: A = [h0 | h1], W = Wcat1
        k_gemm_gates<<<gemmGrid, 256, 0, stream>>>(hbf, hbf + (size_t)NN * HH,
                                                   wcat + (size_t)FOURH * KCAT, gates);
        k_pointwise<<<pwBlocks, 256, 0, stream>>>(gates, bias + FOURH,
                                                  h_base + (size_t)NN * HH,
                                                  c_base + (size_t)NN * HH,
                                                  hbf + (size_t)NN * HH);
    }
}
